// Attention_GRU_66529043415336
// MI455X (gfx1250) — matrix operand order repaired and checked by reading, not compiled, not run on hardware
//
#include <hip/hip_runtime.h>
#include <stdint.h>

#define B_  32
#define T_  512
#define D_  512
#define U_  512
#define K3_ 1536

typedef __attribute__((ext_vector_type(16))) __bf16 v16bf;
typedef __attribute__((ext_vector_type(8)))  __bf16 v8bf;
typedef __attribute__((ext_vector_type(8)))  float  v8f;

// ---------------- CDNA5 async global->LDS copy (ASYNCcnt path) ----------------
__device__ __forceinline__ void async_copy_b128(uint32_t lds_byte_addr, const void* gptr) {
  // per-lane: LDS[lds_byte_addr] <= 16B from MEM[gptr]; tracked by ASYNCcnt
  asm volatile("global_load_async_to_lds_b128 %0, %1, off"
               :: "v"(lds_byte_addr), "v"(gptr)
               : "memory");
}
__device__ __forceinline__ void wait_async0() {
  asm volatile("s_wait_asynccnt 0x0" ::: "memory");
}

// ---------------- WMMA fragment loaders (CDNA5 16-bit layouts) ----------------
// A (16x32, MxK): lane<16 holds row M=lane, K {0..7,16..23}; lane>=16 row M=lane-16, K {8..15,24..31}
__device__ __forceinline__ v16bf load_a_frag(const __bf16* base, int ld, int m0, int kbase, int lane) {
  const int row = m0 + (lane & 15);
  const int sel = (lane >> 4) * 8;
  const __bf16* p = base + (size_t)row * ld + kbase + sel;
  v8bf lo = *(const v8bf*)p;
  v8bf hi = *(const v8bf*)(p + 16);
  v16bf a;
#pragma unroll
  for (int i = 0; i < 8; ++i) { a[i] = lo[i]; a[i + 8] = hi[i]; }
  return a;
}
// B (32x16, KxN) from N-major (transposed) weights, row stride 512:
// lane<16 holds column N=lane, K {0..7,16..23}; lane>=16 column N=lane-16, K {8..15,24..31}
__device__ __forceinline__ v16bf load_b_frag(const __bf16* wt, int n0, int kbase, int lane) {
  const __bf16* p = wt + (size_t)(n0 + (lane & 15)) * 512 + kbase + (lane >> 4) * 8;
  v8bf lo = *(const v8bf*)p;
  v8bf hi = *(const v8bf*)(p + 16);
  v16bf b;
#pragma unroll
  for (int i = 0; i < 8; ++i) { b[i] = lo[i]; b[i + 8] = hi[i]; }
  return b;
}

__device__ __forceinline__ v8f wmma_bf16(v8f c, v16bf a, v16bf b) {
  return __builtin_amdgcn_wmma_f32_16x16x32_bf16(false, a, false, b, (short)0, c, false, false);
}

__device__ __forceinline__ float hsig(float x) {
  return fminf(fmaxf(0.2f * x + 0.5f, 0.0f), 1.0f);
}

// ---------------- device-scope barrier for the persistent scan ----------------
__device__ __forceinline__ void grid_barrier(unsigned* bar, unsigned* epoch, unsigned nblocks) {
  __syncthreads();
  *epoch += 1u;
  if (threadIdx.x == 0) {
    __threadfence();                               // release
    atomicAdd(bar, 1u);
    const unsigned target = (*epoch) * nblocks;
    while (__hip_atomic_load(bar, __ATOMIC_RELAXED, __HIP_MEMORY_SCOPE_AGENT) < target) {
      __builtin_amdgcn_s_sleep(1);
    }
  }
  __syncthreads();
  __builtin_amdgcn_fence(__ATOMIC_ACQUIRE, "agent");  // acquire for all threads
}

// ---------------- prep kernels ----------------
__global__ void k_cvt_x(const float* __restrict__ x, __bf16* __restrict__ xb, int n) {
  int i = blockIdx.x * blockDim.x + threadIdx.x;
  if (i < n) xb[i] = (__bf16)x[i];
}
// w:[512][1536] -> wt:[1536][512] bf16
__global__ void k_transpose_w(const float* __restrict__ w, __bf16* __restrict__ wt) {
  int i = blockIdx.x * blockDim.x + threadIdx.x;
  if (i < K3_ * D_) {
    int n = i / D_, d = i - n * D_;
    wt[i] = (__bf16)w[d * K3_ + n];
  }
}
__global__ void k_init(float* hf, __bf16* hb, unsigned* bar) {
  int i = blockIdx.x * blockDim.x + threadIdx.x;
  if (i < B_ * U_) { hf[i] = 0.0f; hb[i] = (__bf16)0.0f; }
  if (i == 0) *bar = 0u;
}

// ---------------- phase 1: matrix_x = x @ kernel + bias ----------------
// 1x4 register blocking: one A-fragment feeds 4 B-fragments / 4 accumulators.
// xb:[B*T][D], wtk:[K3][D] (N-major), mx stored remapped as [t*B + b][K3]
__global__ void __launch_bounds__(256) k_gemm_mx(const __bf16* __restrict__ xb,
                                                 const __bf16* __restrict__ wtk,
                                                 const float*  __restrict__ bias,
                                                 float* __restrict__ mx) {
  const int lane  = threadIdx.x & 31;
  const int wid   = threadIdx.x >> 5;
  const int wflat = blockIdx.x * 8 + wid;
  const int MT    = (B_ * T_) / 16;     // 1024 row-tiles
  const int mt    = wflat % MT;
  const int ng4   = wflat / MT;         // 0..23 groups of four 16-col tiles
  const int m0 = mt * 16, n0 = ng4 * 64;

  v8f c0 = {}, c1 = {}, c2 = {}, c3 = {};
#pragma unroll 4
  for (int k = 0; k < D_; k += 32) {
    v16bf a  = load_a_frag(xb, D_, m0, k, lane);
    v16bf b0 = load_b_frag(wtk, n0 +  0, k, lane);
    v16bf b1 = load_b_frag(wtk, n0 + 16, k, lane);
    v16bf b2 = load_b_frag(wtk, n0 + 32, k, lane);
    v16bf b3 = load_b_frag(wtk, n0 + 48, k, lane);
    c0 = wmma_bf16(c0, a, b0);
    c1 = wmma_bf16(c1, a, b1);
    c2 = wmma_bf16(c2, a, b2);
    c3 = wmma_bf16(c3, a, b3);
  }

#pragma unroll
  for (int j = 0; j < 4; ++j) {
    v8f c = (j == 0) ? c0 : (j == 1) ? c1 : (j == 2) ? c2 : c3;
    const int nc = n0 + j * 16 + (lane & 15);
    const float bv = bias[nc];
#pragma unroll
    for (int r = 0; r < 8; ++r) {
      int ri = m0 + r + ((lane >> 4) << 3);   // row in [B*T), b-major
      int t  = ri & (T_ - 1);
      int b  = ri >> 9;
      mx[(size_t)(t * B_ + b) * K3_ + nc] = c[r] + bv;
    }
  }
}

// ---------------- phase 2: persistent GRU scan ----------------
// 32 blocks x 256 threads; block g owns z/r columns [g*32, g*32+32) and hh columns [g*16, g*16+16).
// Recurrent weights (N-major bf16) staged once into LDS via async loads.
__global__ void __launch_bounds__(256, 1) k_scan(const float*  __restrict__ mx,   // [T*B][K3]
                                                 const __bf16* __restrict__ wtr,  // [K3][D]
                                                 float* hf, __bf16* hb,
                                                 float* zbuf, __bf16* rhb,
                                                 float* __restrict__ out,         // [B][T][U]
                                                 unsigned* bar) {
  __shared__ __align__(32) __bf16 lds_zr[32 * 512];  // 32KB: this block's z/r weight rows
  __shared__ __align__(32) __bf16 lds_h [16 * 512];  // 16KB: this block's hh weight rows
  const int g    = blockIdx.x;
  const int tid  = threadIdx.x;
  const int lane = tid & 31;
  const int wid  = tid >> 5;

  // ---- stage 48KB of weights into LDS with CDNA5 async copies ----
  {
    const char* base   = (const char*)wtr;
    const uint32_t zr0 = (uint32_t)(uintptr_t)&lds_zr[0];
    const uint32_t h0  = (uint32_t)(uintptr_t)&lds_h[0];
#pragma unroll
    for (int it = 0; it < 12; ++it) {
      int c = tid + it * 256;                     // 3072 x 16B chunks
      if (c < 2048) {
        async_copy_b128(zr0 + (uint32_t)c * 16u,
                        base + (size_t)g * 32768 + (size_t)c * 16);
      } else {
        int c2 = c - 2048;
        async_copy_b128(h0 + (uint32_t)c2 * 16u,
                        base + (size_t)(1024 + g * 16) * 1024 + (size_t)c2 * 16);
      }
    }
    wait_async0();
  }
  __syncthreads();

  unsigned epoch = 0;
  for (int t = 0; t < T_; ++t) {
    // ---- phase A: inner = h @ Wr_zr; compute z (cols<512) / r*h (cols>=512) ----
    if (wid < 4) {
      const int mt = wid & 1, ntl = wid >> 1;
      v8f c = {};
#pragma unroll 4
      for (int k = 0; k < U_; k += 32) {
        v16bf a = load_a_frag(hb, U_, mt * 16, k, lane);
        v16bf b = load_b_frag(lds_zr, ntl * 16, k, lane);
        c = wmma_bf16(c, a, b);
      }
      const int cl = ntl * 16 + (lane & 15);
      const int ng = g * 32 + cl;                 // global column in [0,1024)
#pragma unroll
      for (int r = 0; r < 8; ++r) {
        int m = mt * 16 + r + ((lane >> 4) << 3);
        float mv = mx[(size_t)(t * B_ + m) * K3_ + ng];
        float s  = hsig(mv + c[r]);
        if (ng < U_) {
          zbuf[m * U_ + ng] = s;                  // z gate
        } else {
          int u = ng - U_;                        // r gate -> r*h in bf16
          rhb[m * U_ + u] = (__bf16)(s * hf[m * U_ + u]);
        }
      }
    } else if (t + 1 < T_) {
      // waves 4..7: prefetch next timestep's contiguous mx slab (192KB) toward L2
      const float* nxt = mx + (size_t)((t + 1) * B_) * K3_;
      const int i = (wid - 4) * 32 + lane;        // 0..127
      for (int ofs = i * 64; ofs < B_ * K3_; ofs += 128 * 64) {
        __builtin_prefetch(nxt + ofs, 0, 1);
      }
    }
    grid_barrier(bar, &epoch, 32);

    // ---- phase B: hh = tanh(mx3 + (r*h) @ Wr_h); h_new = z*h + (1-z)*hh ----
    if (wid < 2) {
      const int mt = wid;
      v8f c = {};
#pragma unroll 4
      for (int k = 0; k < U_; k += 32) {
        v16bf a = load_a_frag(rhb, U_, mt * 16, k, lane);
        v16bf b = load_b_frag(lds_h, 0, k, lane);
        c = wmma_bf16(c, a, b);
      }
      const int u = g * 16 + (lane & 15);         // unit in [0,512)
#pragma unroll
      for (int r = 0; r < 8; ++r) {
        int m = mt * 16 + r + ((lane >> 4) << 3);
        float mv  = mx[(size_t)(t * B_ + m) * K3_ + 2 * U_ + u];
        float hhv = tanhf(mv + c[r]);
        float z   = zbuf[m * U_ + u];
        float hp  = hf[m * U_ + u];
        float hn  = z * hp + (1.0f - z) * hhv;
        hf[m * U_ + u] = hn;
        hb[m * U_ + u] = (__bf16)hn;
        out[(size_t)m * (T_ * U_) + (size_t)t * U_ + u] = hn;
      }
    }
    grid_barrier(bar, &epoch, 32);
  }
}

// ---------------- host launcher ----------------
extern "C" void kernel_launch(void* const* d_in, const int* in_sizes, int n_in,
                              void* d_out, int out_size, void* d_ws, size_t ws_size,
                              hipStream_t stream) {
  (void)in_sizes; (void)n_in; (void)out_size; (void)ws_size;
  const float* x    = (const float*)d_in[0];
  const float* wk   = (const float*)d_in[1];
  const float* wr   = (const float*)d_in[2];
  const float* bias = (const float*)d_in[3];
  float* out = (float*)d_out;

  char* p = (char*)d_ws;
  auto take = [&](size_t bytes) -> char* {
    char* q = p;
    p += (bytes + 255) & ~(size_t)255;
    return q;
  };
  __bf16*   xb  = (__bf16*)take((size_t)B_ * T_ * D_ * 2);   // 16 MB
  __bf16*   wtk = (__bf16*)take((size_t)K3_ * D_ * 2);       // 1.5 MB
  __bf16*   wtr = (__bf16*)take((size_t)K3_ * D_ * 2);       // 1.5 MB
  float*    mx  = (float*) take((size_t)B_ * T_ * K3_ * 4);  // 96 MB, [t][b][k]
  float*    hf  = (float*) take((size_t)B_ * U_ * 4);
  __bf16*   hb  = (__bf16*)take((size_t)B_ * U_ * 2);
  float*    zb  = (float*) take((size_t)B_ * U_ * 4);
  __bf16*   rhb = (__bf16*)take((size_t)B_ * U_ * 2);
  unsigned* bar = (unsigned*)take(256);

  const int nx = B_ * T_ * D_;
  k_cvt_x<<<(nx + 255) / 256, 256, 0, stream>>>(x, xb, nx);
  k_transpose_w<<<(K3_ * D_ + 255) / 256, 256, 0, stream>>>(wk, wtk);
  k_transpose_w<<<(K3_ * D_ + 255) / 256, 256, 0, stream>>>(wr, wtr);
  k_init<<<(B_ * U_ + 255) / 256, 256, 0, stream>>>(hf, hb, bar);
  // 1024 row-tiles x 24 four-wide col-groups, 8 waves per block
  k_gemm_mx<<<(B_ * T_ / 16) * (K3_ / 64) / 8, 256, 0, stream>>>(xb, wtk, bias, mx);
  k_scan<<<32, 256, 0, stream>>>(mx, wtr, hf, hb, zb, rhb, out, bar);
}
